// TFLayer_boostrap_calc_woe_77352361001231
// MI455X (gfx1250) — matrix operand order repaired and checked by reading, not compiled, not run on hardware
//
#include <hip/hip_runtime.h>
#include <hip/hip_bf16.h>

typedef __attribute__((ext_vector_type(16))) _Float16 v16h;
typedef __attribute__((ext_vector_type(8)))  float    v8f;

#define FPEPS 0.0001f

// Workspace layout (floats):
//   [0,       FB)     acc_bad    (sum over n of x * (y==1))
//   [FB,     2FB)     acc_good   (sum over n of x * (y==0))
//   [2FB,    3FB)     acc_tot    (sum over n of x)
//   [3FB]             n1 (count of y==1)
//   [3FB+1, 4FB+1)    w_use

__global__ __launch_bounds__(256)
void zero_kernel(float* __restrict__ p, int n) {
    int i = blockIdx.x * blockDim.x + threadIdx.x;
    if (i < n) p[i] = 0.0f;
}

// Phase 1: D(16x16) += A(16x32) * B(32x16) per wave.
//   A = x^T tile: row M = fb column (16-wide tile), K = row index n
//   B = mask:     col 0 = (y==1), col 1 = (y==0), col 2 = 1, cols 3..15 = 0
// f16 inputs exact (0/1 data); f32 accumulation exact (counts < 2^24).
// Hot loop: no guards, branchless mask build from one ballot, y value
// software-pipelined one iteration ahead so its load latency hides behind
// the 16 x-loads + WMMA.
__global__ __launch_bounds__(32)
void woe_reduce_wmma(const float* __restrict__ x, const int* __restrict__ y,
                     float* __restrict__ acc, int Ntot, int FB, int rowsPerChunk)
{
    const int lane = threadIdx.x;     // 0..31
    const int grp  = lane >> 4;       // 0 or 1
    const int lo   = lane & 15;
    const int fb0  = blockIdx.x * 16; // fb tile base
    const int nBeg = blockIdx.y * rowsPerChunk;
    int nEnd = nBeg + rowsPerChunk;
    if (nEnd > Ntot) nEnd = Ntot;

    // per-lane column-role masks (branchless bits select in the loop)
    const unsigned uA = (lo == 0) ? 0xFFFFFFFFu : 0u;   // m1 column
    const unsigned uB = (lo == 1) ? 0xFFFFFFFFu : 0u;   // m0 column
    const unsigned uC = (lo == 2) ? 0xFFFFFFFFu : 0u;   // ones column

    v8f c = {0.f,0.f,0.f,0.f,0.f,0.f,0.f,0.f};
    int n1cnt = 0;

    const float* px = x + nBeg * FB + fb0 + lo;  // per-lane column base
    const int*   py = y + nBeg + lane;           // per-lane y base
    const int    rowStride = 32 * FB;

    const int nFull = nBeg + ((nEnd - nBeg) & ~31);  // end of full chunks

    if (nBeg < nFull) {
        int yv = *py;                              // prime the pipeline
        for (int n0 = nBeg; n0 < nFull; n0 += 32) {
            unsigned m1 = __builtin_amdgcn_ballot_w32(yv == 1);  // uniform
            // branchless prefetch of next chunk's y (clamped offset, always safe)
            int nextOff = (n0 + 64 <= nFull) ? 32 : 0;
            yv = py[nextOff];
            py += 32;

            n1cnt += __popc(m1);
            unsigned bits = (m1 & uA) | (~m1 & uB) | uC;

            v16h a, b;
#pragma unroll
            for (int h = 0; h < 16; ++h) {
                // A slot map (16-bit A 16x32): slot h -> K = ((h<8)?h:h+8) + 8*grp
                int k = ((h < 8) ? h : h + 8) + 8 * grp;
                a[h] = (_Float16)px[k * FB];
                // B slot map (16-bit B 32x16): slot h -> K = ((h<8)?h:h+8) + 8*grp
                b[h] = (_Float16)(float)((bits >> k) & 1u);
            }
            c = __builtin_amdgcn_wmma_f32_16x16x32_f16(
                    false, a, false, b, (short)0, c, false, false);
            px += rowStride;
        }
    }

    // ---- guarded tail: at most one partial 32-row chunk ----
    if (nFull < nEnd) {
        int n0 = nFull;
        int valid = (n0 + lane) < Ntot;
        int yv = valid ? *py : -1;
        unsigned vm  = __builtin_amdgcn_ballot_w32(valid != 0);
        unsigned m1  = __builtin_amdgcn_ballot_w32(yv == 1);
        unsigned m0b = __builtin_amdgcn_ballot_w32(yv == 0);
        n1cnt += __popc(m1);
        unsigned bits = (m1 & uA) | (m0b & uB) | (vm & uC);

        v16h a, b;
#pragma unroll
        for (int h = 0; h < 16; ++h) {
            int k = ((h < 8) ? h : h + 8) + 8 * grp;
            a[h] = (_Float16)(((n0 + k) < Ntot) ? px[k * FB] : 0.0f);
            b[h] = (_Float16)(float)((bits >> k) & 1u);
        }
        c = __builtin_amdgcn_wmma_f32_16x16x32_f16(
                false, a, false, b, (short)0, c, false, false);
    }

    // D layout: lane N = lo, c[r] = row M = r + 8*grp
    if (lo < 3) {
#pragma unroll
        for (int r = 0; r < 8; ++r)
            atomicAdd(&acc[lo * FB + fb0 + r + 8 * grp], c[r]);
    }
    if (blockIdx.x == 0 && lane == 0)
        atomicAdd(&acc[3 * FB], (float)n1cnt);
}

// Phase 2: WOE + WLS fits (tiny: one thread per feature).
__global__ __launch_bounds__(64)
void woe_fit(const float* __restrict__ acc, float* __restrict__ woe_out,
             float* __restrict__ adj_out, float* __restrict__ w_use,
             int Ntot, int F, int B)
{
    int f = threadIdx.x;
    if (f >= F) return;
    const int FB = F * B;
    float n1 = acc[3 * FB];
    float n0 = (float)Ntot - n1;

    float woe[16], tp[16];
    for (int bi = 0; bi < B; ++bi) {
        float bad  = acc[0 * FB + f * B + bi];
        float good = acc[1 * FB + f * B + bi];
        float tot  = acc[2 * FB + f * B + bi];
        woe[bi] = __logf(bad / n1 + FPEPS) - __logf(good / n0 + FPEPS);
        tp[bi]  = tot / (float)Ntot;
        woe_out[f * B + bi] = woe[bi];
    }

    const int K = B - 1;     // 15
    float w[15], pct[15], idxv[15], ind[15];
    float psum = 0.0f;
    for (int k = 0; k < K; ++k) { w[k] = woe[k + 1]; psum += tp[k + 1]; }
    for (int k = 0; k < K; ++k) pct[k] = tp[k + 1] / psum;

    int first = 0;
    for (int k = 0; k < K; ++k) { if (pct[k] != 0.0f) { first = k; break; } }
    for (int k = 0; k < K; ++k) {
        int iv = (k + 1) - first; if (iv < 0) iv = 0;
        idxv[k] = (float)iv;
        ind[k]  = (iv != 0) ? 1.0f : 0.0f;
    }

    // --- WLS fit 1: X1 = [ind, idx], weights pct, target w
    float a00=0,a01=0,a11=0,b0=0,b1=0;
    for (int k = 0; k < K; ++k) {
        float p = pct[k], i1 = ind[k], i2 = idxv[k];
        a00 += p*i1*i1; a01 += p*i1*i2; a11 += p*i2*i2;
        b0  += p*i1*w[k]; b1 += p*i2*w[k];
    }
    float det1 = a00*a11 - a01*a01;
    float c0 = 0.0f, c1 = 0.0f;
    if (__builtin_fabsf(det1) > 1e-12f) {
        c0 = ( a11*b0 - a01*b1) / det1;
        c1 = (-a01*b0 + a00*b1) / det1;
    }
    float fit1[15];
    for (int k = 0; k < K; ++k) fit1[k] = ind[k]*c0 + idxv[k]*c1;

    // --- WLS fit 2: X2 = [ind, idx, idx^2]
    float m00=0,m01=0,m02=0,m11=0,m12=0,m22=0,r0=0,r1v=0,r2v=0;
    for (int k = 0; k < K; ++k) {
        float p = pct[k], i1 = ind[k], i2 = idxv[k], i3 = idxv[k]*idxv[k];
        m00 += p*i1*i1; m01 += p*i1*i2; m02 += p*i1*i3;
        m11 += p*i2*i2; m12 += p*i2*i3; m22 += p*i3*i3;
        r0  += p*i1*w[k]; r1v += p*i2*w[k]; r2v += p*i3*w[k];
    }
    float det2 = m00*(m11*m22 - m12*m12) - m01*(m01*m22 - m12*m02)
               + m02*(m01*m12 - m11*m02);
    float d0 = 0.0f, d1 = 0.0f, d2 = 0.0f;
    if (__builtin_fabsf(det2) > 1e-12f) {
        float i00 = (m11*m22 - m12*m12) / det2;
        float i01 = (m02*m12 - m01*m22) / det2;
        float i02 = (m01*m12 - m02*m11) / det2;
        float i11 = (m00*m22 - m02*m02) / det2;
        float i12 = (m01*m02 - m00*m12) / det2;
        float i22 = (m00*m11 - m01*m01) / det2;
        d0 = i00*r0 + i01*r1v + i02*r2v;
        d1 = i01*r0 + i11*r1v + i12*r2v;
        d2 = i02*r0 + i12*r1v + i22*r2v;
    }
    float fit2[15];
    for (int k = 0; k < K; ++k)
        fit2[k] = ind[k]*d0 + idxv[k]*d1 + idxv[k]*idxv[k]*d2;

    // --- R^2 blend
    float mean_w = 0.0f;
    for (int k = 0; k < K; ++k) mean_w += w[k]*pct[k];
    float sst = 0.0f, e1 = 0.0f, e2 = 0.0f;
    for (int k = 0; k < K; ++k) {
        float dw = w[k] - mean_w;
        sst += dw*dw*pct[k];
        float q1 = fit1[k] - w[k]; e1 += q1*q1*pct[k];
        float q2 = fit2[k] - w[k]; e2 += q2*q2*pct[k];
    }
    float rr1 = 1.0f - e1/sst;
    float rr2 = 1.0f - e2/sst;

    adj_out[f * B + 0] = woe[0];
    w_use[f * B + 0]   = (woe[0] != woe[0]) ? 0.0f : woe[0];
    for (int k = 0; k < K; ++k) {
        float adj = (fit1[k]*rr1 + fit2[k]*rr2) / (rr1 + rr2);
        adj_out[f * B + 1 + k] = adj;
        w_use[f * B + 1 + k]   = (adj != adj) ? 0.0f : adj;
    }
}

// Phase 3: rt[n][f] = sum_b (x[n,f,b]==1) * w_use[f,b]. Pure bandwidth.
__global__ __launch_bounds__(256)
void rt_kernel(const float* __restrict__ x, const float* __restrict__ w_use,
               float* __restrict__ rt, int total, int F, int B)
{
    __shared__ float ws[800];
    const int FB = F * B;
    for (int i = threadIdx.x; i < FB; i += blockDim.x) ws[i] = w_use[i];
    __syncthreads();

    int idx = blockIdx.x * blockDim.x + threadIdx.x;   // total = 5e6 < 2^31
    if (idx >= total) return;
    int f = idx % F;
    int n = idx / F;

    const float* xr = x + (n * F + f) * B;   // contiguous 64B segment
    const float* wf = &ws[f * 16];
    float s = 0.0f;
#pragma unroll
    for (int bi = 0; bi < 16; bi += 4) {
        float4 v = *(const float4*)(xr + bi);
        s += (v.x == 1.0f ? wf[bi + 0] : 0.0f)
           + (v.y == 1.0f ? wf[bi + 1] : 0.0f)
           + (v.z == 1.0f ? wf[bi + 2] : 0.0f)
           + (v.w == 1.0f ? wf[bi + 3] : 0.0f);
    }
    rt[idx] = s;
}

extern "C" void kernel_launch(void* const* d_in, const int* in_sizes, int n_in,
                              void* d_out, int out_size, void* d_ws, size_t ws_size,
                              hipStream_t stream)
{
    (void)n_in; (void)out_size; (void)ws_size;
    const float* x = (const float*)d_in[0];
    const int*   y = (const int*)d_in[1];

    const int Ntot = in_sizes[1];            // 100000
    const int FB   = in_sizes[0] / Ntot;     // 800
    const int B    = 16;
    const int F    = FB / B;                 // 50

    float* acc    = (float*)d_ws;            // 3*FB + 1 floats
    float* w_use  = acc + 3 * FB + 1;        // FB floats

    float* rt      = (float*)d_out;
    float* woe_out = rt + (long long)Ntot * F;
    float* adj_out = woe_out + FB;

    // zero accumulators (atomics below; must re-zero every call)
    int zn = 3 * FB + 1;
    zero_kernel<<<(zn + 255) / 256, 256, 0, stream>>>(acc, zn);

    // phase 1: mask-GEMM reduction via WMMA
    const int rowsPerChunk = 800;            // 25 WMMA k-steps per block
    int nChunks = (Ntot + rowsPerChunk - 1) / rowsPerChunk;
    dim3 g1(FB / 16, nChunks);
    woe_reduce_wmma<<<g1, 32, 0, stream>>>(x, y, acc, Ntot, FB, rowsPerChunk);

    // phase 2: WOE + WLS (tiny)
    woe_fit<<<1, 64, 0, stream>>>(acc, woe_out, adj_out, w_use, Ntot, F, B);

    // phase 3: rt pass
    int total = Ntot * F;
    int blocks = (total + 255) / 256;
    rt_kernel<<<blocks, 256, 0, stream>>>(x, w_use, rt, total, F, B);
}
